// U_CRF_35527969472872
// MI455X (gfx1250) — matrix operand order repaired and checked by reading, not compiled, not run on hardware
//
#include <hip/hip_runtime.h>
#include <hip/hip_bf16.h>

// ---------- types for WMMA ----------
typedef __attribute__((ext_vector_type(16))) __bf16 v16bf;
typedef __attribute__((ext_vector_type(8)))  __bf16 bf16x8;
typedef __attribute__((ext_vector_type(8)))  float  v8f;

union Frag16 { v16bf v; bf16x8 h[2]; };

__device__ __forceinline__ __bf16 f2bf(float f) {
    unsigned u = __builtin_bit_cast(unsigned, f);
    u += 0x7FFFu + ((u >> 16) & 1u);            // round-to-nearest-even
    unsigned short h = (unsigned short)(u >> 16);
    return __builtin_bit_cast(__bf16, h);
}
__device__ __forceinline__ unsigned pack2bf(float a, float b) {
    unsigned ua = __builtin_bit_cast(unsigned, a);
    ua += 0x7FFFu + ((ua >> 16) & 1u);
    unsigned ub = __builtin_bit_cast(unsigned, b);
    ub += 0x7FFFu + ((ub >> 16) & 1u);
    return (ua >> 16) | (ub & 0xFFFF0000u);
}

#define BM 128
#define BN 128
#define BK 64
#define LKP 72                 // LDS row pitch in bf16: 144B = 9*16B, conflict-free b128 reads
#define NEG_BIG -3.402823466e38f

// ---------- generic bf16-WMMA GEMM ----------
// C[M,N] (f32) = A[M,K] (f32) @ op(B)    op(B)=B[N,K]^T if BT else B[K,N]
// f32 -> bf16 conversion during vectorized LDS staging (float4 loads, b64 LDS stores).
// 256 threads = 8 waves (4 along M x 2 along N); each wave computes 32x64 (2x4 frags).
template<bool BT>
__global__ __launch_bounds__(256) void gemm_bf16_k(
    const float* __restrict__ A, const float* __restrict__ B,
    float* __restrict__ C, int M, int N, int K)
{
    __shared__ __bf16 lsA[BM * LKP];   // [m][k]
    __shared__ __bf16 lsB[BN * LKP];   // [n][k] (transposed during staging for NN)

    const int tid  = threadIdx.x;
    const int lane = tid & 31;
    const int wave = tid >> 5;
    const int wm   = wave & 3;         // 4 waves along M (32 rows each)
    const int wn   = wave >> 2;        // 2 waves along N (64 cols each)
    const int m0   = blockIdx.y * BM;
    const int n0   = blockIdx.x * BN;
    const int half = lane >> 4;        // 0: lanes 0-15, 1: lanes 16-31
    const int r    = lane & 15;

    v8f acc[2][4] = {};

    for (int kb = 0; kb < K; kb += BK) {
        // ---- stage A tile: 128x64 f32 -> bf16 (8 float4 per thread) ----
        #pragma unroll
        for (int i = 0; i < (BM * BK) / (4 * 256); ++i) {
            int idx = tid + i * 256;               // float4 slot
            int m = idx >> 4, k4 = (idx & 15) << 2;
            const float4 v = *reinterpret_cast<const float4*>(
                &A[(size_t)(m0 + m) * K + kb + k4]);
            uint2 pk = { pack2bf(v.x, v.y), pack2bf(v.z, v.w) };
            *reinterpret_cast<uint2*>(&lsA[m * LKP + k4]) = pk;
        }
        // ---- stage B tile: 128x64 -> LDS [n][k] ----
        if (BT) {
            #pragma unroll
            for (int i = 0; i < (BN * BK) / (4 * 256); ++i) {
                int idx = tid + i * 256;
                int n = idx >> 4, k4 = (idx & 15) << 2;
                const float4 v = *reinterpret_cast<const float4*>(
                    &B[(size_t)(n0 + n) * K + kb + k4]);
                uint2 pk = { pack2bf(v.x, v.y), pack2bf(v.z, v.w) };
                *reinterpret_cast<uint2*>(&lsB[n * LKP + k4]) = pk;
            }
        } else {
            #pragma unroll
            for (int i = 0; i < (BN * BK) / (4 * 256); ++i) {
                int idx = tid + i * 256;
                int k = idx >> 5, n4 = (idx & 31) << 2;  // coalesced read along n
                const float4 v = *reinterpret_cast<const float4*>(
                    &B[(size_t)(kb + k) * N + n0 + n4]);
                lsB[(n4 + 0) * LKP + k] = f2bf(v.x);
                lsB[(n4 + 1) * LKP + k] = f2bf(v.y);
                lsB[(n4 + 2) * LKP + k] = f2bf(v.z);
                lsB[(n4 + 3) * LKP + k] = f2bf(v.w);
            }
        }
        if (kb + BK < K) {  // hint next K tile (global_prefetch_b8)
            __builtin_prefetch(&A[(size_t)(m0 + (tid >> 1)) * K + kb + BK + (tid & 1) * 32], 0, 1);
            if (!BT) __builtin_prefetch(&B[(size_t)(kb + BK + (tid >> 2)) * N + n0 + (tid & 3) * 32], 0, 1);
        }
        __syncthreads();

        // ---- 2x4 WMMA fragments per wave, two 32-deep k-steps per stage ----
        #pragma unroll
        for (int kk = 0; kk < BK; kk += 32) {
            Frag16 a[2];
            #pragma unroll
            for (int im = 0; im < 2; ++im) {
                const int arow = 32 * wm + 16 * im + r;
                // A 16x32 bf16: lanes 0-15 -> K{0..7,16..23}; lanes 16-31 -> K{8..15,24..31}
                a[im].h[0] = *reinterpret_cast<const bf16x8*>(&lsA[arow * LKP + kk + half * 8]);
                a[im].h[1] = *reinterpret_cast<const bf16x8*>(&lsA[arow * LKP + kk + half * 8 + 16]);
            }
            #pragma unroll
            for (int in = 0; in < 4; ++in) {
                Frag16 b;
                const int brow = 64 * wn + 16 * in + r;
                // B 32x16 bf16: lanes 0-15 -> K{0..7,16..23}; lanes 16-31 -> K{8..15,24..31} (N = lane&15)
                b.h[0] = *reinterpret_cast<const bf16x8*>(&lsB[brow * LKP + kk + half * 8]);
                b.h[1] = *reinterpret_cast<const bf16x8*>(&lsB[brow * LKP + kk + half * 8 + 16]);
                #pragma unroll
                for (int im = 0; im < 2; ++im)
                    acc[im][in] = __builtin_amdgcn_wmma_f32_16x16x32_bf16(
                        false, a[im].v, false, b.v, (short)0, acc[im][in], false, false);
            }
        }
        __syncthreads();
    }

    // ---- epilogue: C/D layout: lane -> (N = lane&15, M = vgpr + 8*(lane>>4)) ----
    #pragma unroll
    for (int im = 0; im < 2; ++im)
        #pragma unroll
        for (int in = 0; in < 4; ++in) {
            const int col  = n0 + 64 * wn + 16 * in + r;
            const int mrow = m0 + 32 * wm + 16 * im + half * 8;
            #pragma unroll
            for (int e = 0; e < 8; ++e)
                C[(size_t)(mrow + e) * N + col] = acc[im][in][e];
        }
}

// ---------- utility kernels ----------
__global__ __launch_bounds__(256) void fill_k(float* p, int n, float v) {
    int i = blockIdx.x * 256 + threadIdx.x;
    if (i < n) p[i] = v;
}

// edge embedding mean + scatter-add onto src/trg nodes (float4 per thread)
__global__ __launch_bounds__(256) void edge_scatter_k(
    const int* __restrict__ ei, const float* __restrict__ tab,
    float* __restrict__ srcE, float* __restrict__ trgE, int E, int D)
{
    int t = blockIdx.x * 256 + threadIdx.x;
    int per = D >> 2;                      // float4 chunks per edge
    if (t >= E * per) return;
    int e = t / per, q = (t - e * per) << 2;
    int s = ei[e], g = ei[E + e];
    const float4 a = *reinterpret_cast<const float4*>(&tab[(size_t)s * D + q]);
    const float4 b = *reinterpret_cast<const float4*>(&tab[(size_t)g * D + q]);
    float m0 = 0.5f * (a.x + b.x), m1 = 0.5f * (a.y + b.y);
    float m2 = 0.5f * (a.z + b.z), m3 = 0.5f * (a.w + b.w);
    atomicAdd(&srcE[(size_t)s * D + q + 0], m0);
    atomicAdd(&srcE[(size_t)s * D + q + 1], m1);
    atomicAdd(&srcE[(size_t)s * D + q + 2], m2);
    atomicAdd(&srcE[(size_t)s * D + q + 3], m3);
    atomicAdd(&trgE[(size_t)g * D + q + 0], m0);
    atomicAdd(&trgE[(size_t)g * D + q + 1], m1);
    atomicAdd(&trgE[(size_t)g * D + q + 2], m2);
    atomicAdd(&trgE[(size_t)g * D + q + 3], m3);
}

__device__ __forceinline__ float wave_max(float v) {
    #pragma unroll
    for (int s = 16; s; s >>= 1) v = fmaxf(v, __shfl_xor(v, s, 32));
    return v;
}
__device__ __forceinline__ float wave_sum(float v) {
    #pragma unroll
    for (int s = 16; s; s >>= 1) v += __shfl_xor(v, s, 32);
    return v;
}

// row-wise softmax over 4096 cols, optionally after (v - gmax) + leaky_relu
__global__ __launch_bounds__(256) void row_softmax_k(
    float* __restrict__ Mt, int cols, const float* __restrict__ gmaxp,
    float slope, int do_lrelu)
{
    __shared__ float red[8];
    __shared__ float bc;
    const size_t base = (size_t)blockIdx.x * cols;
    const float gm = do_lrelu ? gmaxp[0] : 0.0f;

    float vals[16];                       // cols == 4096, 256 threads
    float m = NEG_BIG;
    #pragma unroll
    for (int i = 0; i < 16; ++i) {
        float v = Mt[base + threadIdx.x + i * 256];
        if (do_lrelu) { v -= gm; v = v > 0.0f ? v : v * slope; }
        vals[i] = v; m = fmaxf(m, v);
    }
    m = wave_max(m);
    if ((threadIdx.x & 31) == 0) red[threadIdx.x >> 5] = m;
    __syncthreads();
    if (threadIdx.x == 0) {
        float r0 = red[0];
        for (int i = 1; i < 8; ++i) r0 = fmaxf(r0, red[i]);
        bc = r0;
    }
    __syncthreads();
    const float rmax = bc;
    __syncthreads();

    float s = 0.0f;
    #pragma unroll
    for (int i = 0; i < 16; ++i) { vals[i] = __expf(vals[i] - rmax); s += vals[i]; }
    s = wave_sum(s);
    if ((threadIdx.x & 31) == 0) red[threadIdx.x >> 5] = s;
    __syncthreads();
    if (threadIdx.x == 0) {
        float r0 = 0.0f;
        for (int i = 0; i < 8; ++i) r0 += red[i];
        bc = r0;
    }
    __syncthreads();
    const float inv = 1.0f / bc;
    #pragma unroll
    for (int i = 0; i < 16; ++i)
        Mt[base + threadIdx.x + i * 256] = vals[i] * inv;
}

__global__ __launch_bounds__(256) void rowmax_k(const float* __restrict__ Mt,
                                                float* __restrict__ out, int cols) {
    __shared__ float red[8];
    float m = NEG_BIG;
    const size_t base = (size_t)blockIdx.x * cols;
    for (int c = threadIdx.x; c < cols; c += 256) m = fmaxf(m, Mt[base + c]);
    m = wave_max(m);
    if ((threadIdx.x & 31) == 0) red[threadIdx.x >> 5] = m;
    __syncthreads();
    if (threadIdx.x == 0) {
        float r0 = red[0];
        for (int i = 1; i < 8; ++i) r0 = fmaxf(r0, red[i]);
        out[blockIdx.x] = r0;
    }
}

__global__ __launch_bounds__(256) void reduce_max_k(const float* __restrict__ in,
                                                    float* __restrict__ out, int n) {
    __shared__ float red[8];
    float m = NEG_BIG;
    for (int i = threadIdx.x; i < n; i += 256) m = fmaxf(m, in[i]);
    m = wave_max(m);
    if ((threadIdx.x & 31) == 0) red[threadIdx.x >> 5] = m;
    __syncthreads();
    if (threadIdx.x == 0) {
        float r0 = red[0];
        for (int i = 1; i < 8; ++i) r0 = fmaxf(r0, red[i]);
        out[0] = r0;
    }
}

__global__ __launch_bounds__(256) void rowsum_k(const float* __restrict__ Mt,
                                                float* __restrict__ out, int cols) {
    __shared__ float red[8];
    float s = 0.0f;
    const size_t base = (size_t)blockIdx.x * cols;
    for (int c = threadIdx.x; c < cols; c += 256) s += Mt[base + c];
    s = wave_sum(s);
    if ((threadIdx.x & 31) == 0) red[threadIdx.x >> 5] = s;
    __syncthreads();
    if (threadIdx.x == 0) {
        float r0 = 0.0f;
        for (int i = 0; i < 8; ++i) r0 += red[i];
        out[blockIdx.x] = r0;
    }
}

__global__ __launch_bounds__(256) void transpose_k(const float* __restrict__ in,
                                                   float* __restrict__ out, int R, int C) {
    size_t t = (size_t)blockIdx.x * 256 + threadIdx.x;
    if (t >= (size_t)R * C) return;
    int rr = (int)(t / C), cc = (int)(t - (size_t)rr * C);
    out[(size_t)cc * R + rr] = in[t];
}

// out = (b*x + a*mm) / (b + a*normalize[row]),  a=exp(alpha), b=exp(beta)
__global__ __launch_bounds__(256) void iter_k(
    const float* __restrict__ x, const float* __restrict__ mm,
    const float* __restrict__ normv, const float* __restrict__ alpha,
    const float* __restrict__ beta, float* __restrict__ out, int total, int D)
{
    int i = blockIdx.x * 256 + threadIdx.x;
    if (i >= total) return;
    const float a = __expf(alpha[0]);
    const float b = __expf(beta[0]);
    const int row = i / D;
    out[i] = (b * x[i] + a * mm[i]) / (b + normv[row] * a);
}

// ---------- pipeline ----------
extern "C" void kernel_launch(void* const* d_in, const int* in_sizes, int n_in,
                              void* d_out, int out_size, void* d_ws, size_t ws_size,
                              hipStream_t stream)
{
    (void)n_in; (void)out_size; (void)ws_size;
    const float* x     = (const float*)d_in[0];
    const float* adj   = (const float*)d_in[1];
    const int*   ei    = (const int*)  d_in[2];
    /* d_in[3] node_wgt: computed-but-unused in reference */
    const float* tab   = (const float*)d_in[4];
    const float* Wsrc  = (const float*)d_in[5];
    const float* Wtrg  = (const float*)d_in[6];
    const float* alpha = (const float*)d_in[7];
    const float* beta  = (const float*)d_in[8];

    const int N = 4096, D = 256;
    const int E = in_sizes[2] / 2;

    // workspace carve-up (~100 MB)
    float* p    = (float*)d_ws;
    float* srcE = p; p += (size_t)N * D;
    float* trgE = p; p += (size_t)N * D;
    float* M1   = p; p += (size_t)N * N;   // S -> edge_mat -> logits -> similarity
    float* Tbuf = p; p += (size_t)D * N;   // x2^T @ edge_mat
    float* x1   = p; p += (size_t)N * D;
    float* x2   = p; p += (size_t)N * D;
    float* x2t  = p; p += (size_t)D * N;
    float* tmp  = p; p += (size_t)N * D;   // sim @ out
    float* mm   = p; p += (size_t)N * D;   // adj @ tmp
    float* outA = p; p += (size_t)N * D;
    float* normv= p; p += N;
    float* rowr = p; p += N;
    float* gmax = p; p += 1;

    // 1) zero scatter targets (srcE,trgE contiguous), 2) edge scatter
    fill_k<<<(2 * N * D + 255) / 256, 256, 0, stream>>>(srcE, 2 * N * D, 0.0f);
    edge_scatter_k<<<(E * (D / 4) + 255) / 256, 256, 0, stream>>>(ei, tab, srcE, trgE, E, D);

    // 3) S = srcE @ trgE^T  (4096x4096, K=256); 4) edge_mat = row-softmax(S)
    gemm_bf16_k<true ><<<dim3(N / BN, N / BM), 256, 0, stream>>>(srcE, trgE, M1, N, N, D);
    row_softmax_k<<<N, 256, 0, stream>>>(M1, N, nullptr, 1.0f, 0);

    // 5) x1 = x@Wsrc^T, x2 = x@Wtrg^T  (4096x256, K=256)
    gemm_bf16_k<true ><<<dim3(D / BN, N / BM), 256, 0, stream>>>(x, Wsrc, x1, N, D, D);
    gemm_bf16_k<true ><<<dim3(D / BN, N / BM), 256, 0, stream>>>(x, Wtrg, x2, N, D, D);

    // 6-8) logits = x1 @ (x2^T @ edge_mat)   [associativity kills the 4096^3 GEMM]
    transpose_k<<<(int)(((size_t)N * D + 255) / 256), 256, 0, stream>>>(x2, x2t, N, D);
    gemm_bf16_k<false><<<dim3(N / BN, D / BM), 256, 0, stream>>>(x2t, M1, Tbuf, D, N, N);
    gemm_bf16_k<false><<<dim3(N / BN, N / BM), 256, 0, stream>>>(x1, Tbuf, M1, N, N, D);

    // 9-10) similarity = row-softmax(leaky_relu(logits - global_max))
    rowmax_k<<<N, 256, 0, stream>>>(M1, rowr, N);
    reduce_max_k<<<1, 256, 0, stream>>>(rowr, gmax, N);
    row_softmax_k<<<N, 256, 0, stream>>>(M1, N, gmax, 0.2f, 1);

    // 11) normalize = support.sum(1) = adj.sum(1)  [softmax rows sum to 1]
    rowsum_k<<<N, 256, 0, stream>>>(adj, normv, N);

    // 12) iteration 1: support@x = adj@(sim@x)
    gemm_bf16_k<false><<<dim3(D / BN, N / BM), 256, 0, stream>>>(M1, x, tmp, N, D, N);
    gemm_bf16_k<false><<<dim3(D / BN, N / BM), 256, 0, stream>>>(adj, tmp, mm, N, D, N);
    iter_k<<<(N * D + 255) / 256, 256, 0, stream>>>(x, mm, normv, alpha, beta, outA, N * D, D);

    // 13) iteration 2 -> d_out
    gemm_bf16_k<false><<<dim3(D / BN, N / BM), 256, 0, stream>>>(M1, outA, tmp, N, D, N);
    gemm_bf16_k<false><<<dim3(D / BN, N / BM), 256, 0, stream>>>(adj, tmp, mm, N, D, N);
    iter_k<<<(N * D + 255) / 256, 256, 0, stream>>>(x, mm, normv, alpha, beta, (float*)d_out, N * D, D);
}
